// MultiHeadAttention_89876485636546
// MI455X (gfx1250) — matrix operand order repaired and checked by reading, not compiled, not run on hardware
//
#include <hip/hip_runtime.h>
#include <hip/hip_bf16.h>

// MHA forward for MI455X (gfx1250, wave32, WMMA + async-to-LDS).
// Bandwidth-bound on the mandatory attn materialization (537 MB f32);
// all GEMMs use v_wmma_f32_16x16x32_bf16 with f32 accumulation.
// bf16-source tiles are staged with global_load_async_to_lds_b128 (ASYNCcnt);
// f32-source tiles via vectorized register staging, with sched_barrier fences
// so all staging loads issue before the WMMA block and convert after it.

typedef __bf16 bf16_t;
typedef __attribute__((ext_vector_type(16))) __bf16 v16bf;
typedef __attribute__((ext_vector_type(8)))  float  v8f;
typedef __attribute__((ext_vector_type(4)))  float  f32x4;
typedef __attribute__((ext_vector_type(2)))  unsigned int u32x2;

union BF16Frag { v16bf v; bf16_t e[16]; };
union F32Acc   { v8f   v; float  e[8]; };
union BF16Pack { bf16_t e[4]; u32x2 u; };

#define BATCH 2
#define SEQ   2048
#define DMODEL 1024
#define NHEAD 16
#define DK    64
#define LDSK  48   // padded K stride in LDS (bf16 elems); 96B rows, 16B-aligned

#define WAIT_ASYNC()    asm volatile("s_wait_asynccnt 0x0" ::: "memory")
#define SCHED_FENCE()   __builtin_amdgcn_sched_barrier(0)

// ---------------------------------------------------------------------------
// Async global->LDS tile copy for bf16 sources (no conversion needed).
// Each thread moves 16 B (8 bf16) per step via global_load_async_to_lds_b128.
// Tracked by ASYNCcnt; caller fences with WAIT_ASYNC() before the barrier.
// ---------------------------------------------------------------------------
template <int ROWS>
__device__ inline void tile_async_bf16(bf16_t* __restrict__ dst,
                                       const bf16_t* __restrict__ src,
                                       long ld, int tid) {
#pragma unroll
  for (int i = 0; i < ROWS / 64; ++i) {
    const int e = (i * 256 + tid) * 8;
    const int r = e >> 5, c = e & 31;
    const unsigned lds = (unsigned)(unsigned long long)(dst + r * LDSK + c);
    const bf16_t* g = src + (long)r * ld + c;
    asm volatile("global_load_async_to_lds_b128 %0, %1, off"
                 :: "v"(lds), "v"(g) : "memory");
  }
}

// ---------------------------------------------------------------------------
// Vectorized register staging for f32 sources (converted to bf16 into LDS).
// Each thread moves 4 consecutive elements per step.  ITER = ROWS/32.
// ---------------------------------------------------------------------------
template <int ROWS>
__device__ inline void tile_load_f32(f32x4* reg, const float* __restrict__ src,
                                     long ld, int tid) {
#pragma unroll
  for (int i = 0; i < ROWS / 32; ++i) {
    const int e = (i * 256 + tid) * 4;
    const int r = e >> 5, c = e & 31;
    reg[i] = *(const f32x4*)(src + (long)r * ld + c);
  }
}

template <int ROWS>
__device__ inline void tile_store_f32(bf16_t* __restrict__ dst,
                                      const f32x4* reg, int tid) {
#pragma unroll
  for (int i = 0; i < ROWS / 32; ++i) {
    const int e = (i * 256 + tid) * 4;
    const int r = e >> 5, c = e & 31;
    BF16Pack p;
    p.e[0] = (bf16_t)reg[i].x; p.e[1] = (bf16_t)reg[i].y;
    p.e[2] = (bf16_t)reg[i].z; p.e[3] = (bf16_t)reg[i].w;
    *(u32x2*)(dst + r * LDSK + c) = p.u;
  }
}

// ---------------------------------------------------------------------------
// One K=32 WMMA step for a 32x64 per-wave stripe (2 M-tiles x 4 N-tiles).
// btile[n][k] holds B[k][n] -- the transposed-B pattern used by all GEMMs
// here: W[n,k], Kh[n,k], Vt[n,k].  Layouts per CDNA5 ISA 7.12.2.
// ---------------------------------------------------------------------------
__device__ inline void wave_mma_step(const bf16_t* __restrict__ atile,
                                     const bf16_t* __restrict__ btile,
                                     int waveM, int waveN, int lane,
                                     v8f acc[2][4]) {
  const int lr = lane & 15;
  const int hh = lane >> 4;
  BF16Frag af[2];
#pragma unroll
  for (int mt = 0; mt < 2; ++mt) {
    const bf16_t* row = atile + (waveM + mt * 16 + lr) * LDSK;
#pragma unroll
    for (int v = 0; v < 8; ++v) {
      const int k0 = ((v < 4) ? 0 : 16) + hh * 8 + (v & 3) * 2;
      af[mt].e[2 * v]     = row[k0];
      af[mt].e[2 * v + 1] = row[k0 + 1];
    }
  }
#pragma unroll
  for (int nt = 0; nt < 4; ++nt) {
    BF16Frag bfr;
    const bf16_t* row = btile + (waveN + nt * 16 + lr) * LDSK;
#pragma unroll
    for (int v = 0; v < 8; ++v) {
      const int k0 = ((v < 4) ? 0 : 16) + hh * 8 + (v & 3) * 2;
      bfr.e[2 * v]     = row[k0];
      bfr.e[2 * v + 1] = row[k0 + 1];
    }
#pragma unroll
    for (int mt = 0; mt < 2; ++mt) {
      acc[mt][nt] = __builtin_amdgcn_wmma_f32_16x16x32_bf16(
          false, af[mt].v, false, bfr.v, (short)0, acc[mt][nt], false, false);
    }
  }
}

// ---------------------------------------------------------------------------
// K1: fused QKV projection.  Y = X @ W^T + b  (M=4096, N=1024, K=1024)
// grid = (8, 32, 3); proj 0->Qh [b,h,s,dk], 1->Kh [b,h,s,dk], 2->Vt [b,h,dk,s]
// ---------------------------------------------------------------------------
__global__ __launch_bounds__(256) void mha_proj_kernel(
    const float* __restrict__ xq, const float* __restrict__ xk,
    const float* __restrict__ xv,
    const float* __restrict__ wq, const float* __restrict__ bq,
    const float* __restrict__ wk, const float* __restrict__ bk,
    const float* __restrict__ wv, const float* __restrict__ bv,
    bf16_t* __restrict__ Qh, bf16_t* __restrict__ Kh, bf16_t* __restrict__ Vt) {
  __shared__ bf16_t at[2][128 * LDSK];
  __shared__ bf16_t bt[2][128 * LDSK];
  const int proj = blockIdx.z;
  const float* x    = (proj == 0) ? xq : (proj == 1) ? xk : xv;
  const float* w    = (proj == 0) ? wq : (proj == 1) ? wk : wv;
  const float* bias = (proj == 0) ? bq : (proj == 1) ? bk : bv;
  const int mBase = blockIdx.y * 128;
  const int nBase = blockIdx.x * 128;
  const int tid = threadIdx.x;
  const int wave = tid >> 5, lane = tid & 31;
  const int waveM = (wave & 3) * 32, waveN = (wave >> 2) * 64;
  const float* aSrc = x + (long)mBase * DMODEL;
  const float* bSrc = w + (long)nBase * DMODEL;

  v8f acc[2][4] = {};
  f32x4 aReg[4], bReg[4];
  tile_load_f32<128>(aReg, aSrc, DMODEL, tid);
  tile_load_f32<128>(bReg, bSrc, DMODEL, tid);
  tile_store_f32<128>(at[0], aReg, tid);
  tile_store_f32<128>(bt[0], bReg, tid);
  __syncthreads();
  int cur = 0;
  for (int k0 = 32; k0 < DMODEL; k0 += 32) {
    tile_load_f32<128>(aReg, aSrc + k0, DMODEL, tid);
    tile_load_f32<128>(bReg, bSrc + k0, DMODEL, tid);
    SCHED_FENCE();   // all staging loads issued before the WMMA block
    wave_mma_step(at[cur], bt[cur], waveM, waveN, lane, acc);
    SCHED_FENCE();   // convert/ds_store only after the WMMA block
    tile_store_f32<128>(at[cur ^ 1], aReg, tid);
    tile_store_f32<128>(bt[cur ^ 1], bReg, tid);
    __syncthreads();
    cur ^= 1;
  }
  wave_mma_step(at[cur], bt[cur], waveM, waveN, lane, acc);

  const int lr = lane & 15, hh = lane >> 4;
#pragma unroll
  for (int mt = 0; mt < 2; ++mt)
#pragma unroll
    for (int nt = 0; nt < 4; ++nt) {
      F32Acc u; u.v = acc[mt][nt];
#pragma unroll
      for (int r = 0; r < 8; ++r) {
        const int m = mBase + waveM + mt * 16 + hh * 8 + r;
        const int n = nBase + waveN + nt * 16 + lr;
        const float val = u.e[r] + bias[n];
        const int batch = m >> 11, s = m & (SEQ - 1);
        const int h = n >> 6, d = n & (DK - 1);
        const long hoff = (long)(batch * NHEAD + h);
        if (proj == 0)      Qh[(hoff * SEQ + s) * DK + d] = (bf16_t)val;
        else if (proj == 1) Kh[(hoff * SEQ + s) * DK + d] = (bf16_t)val;
        else                Vt[(hoff * DK + d) * SEQ + s] = (bf16_t)val;
      }
    }
}

// ---------------------------------------------------------------------------
// K2: raw scores = (Qh @ Kh^T) / sqrt(dk) per (b,h).  M=N=2048, K=64.
// grid = (16, 16, 32).  Both tiles staged fully asynchronously (ASYNCcnt);
// NT-stores f32 scores into the attn output region.
// ---------------------------------------------------------------------------
__global__ __launch_bounds__(256) void mha_scores_kernel(
    const bf16_t* __restrict__ Qh, const bf16_t* __restrict__ Kh,
    float* __restrict__ attn) {
  __shared__ bf16_t at[2][128 * LDSK];
  __shared__ bf16_t bt[2][128 * LDSK];
  const int bh = blockIdx.z;
  const int mBase = blockIdx.y * 128;
  const int nBase = blockIdx.x * 128;
  const bf16_t* aSrc = Qh + (long)bh * SEQ * DK + (long)mBase * DK;
  const bf16_t* bSrc = Kh + (long)bh * SEQ * DK + (long)nBase * DK;
  const int tid = threadIdx.x;
  const int wave = tid >> 5, lane = tid & 31;
  const int waveM = (wave & 3) * 32, waveN = (wave >> 2) * 64;

  v8f acc[2][4] = {};
  tile_async_bf16<128>(at[0], aSrc, DK, tid);
  tile_async_bf16<128>(bt[0], bSrc, DK, tid);
  tile_async_bf16<128>(at[1], aSrc + 32, DK, tid);
  tile_async_bf16<128>(bt[1], bSrc + 32, DK, tid);
  WAIT_ASYNC();
  __syncthreads();
  wave_mma_step(at[0], bt[0], waveM, waveN, lane, acc);
  wave_mma_step(at[1], bt[1], waveM, waveN, lane, acc);

  const int lr = lane & 15, hh = lane >> 4;
  float* arow = attn + (long)bh * SEQ * SEQ;
#pragma unroll
  for (int mt = 0; mt < 2; ++mt)
#pragma unroll
    for (int nt = 0; nt < 4; ++nt) {
      F32Acc u; u.v = acc[mt][nt];
#pragma unroll
      for (int r = 0; r < 8; ++r) {
        const int m = mBase + waveM + mt * 16 + hh * 8 + r;
        const int n = nBase + waveN + nt * 16 + lr;
        __builtin_nontemporal_store(u.e[r] * 0.125f,
                                    arow + (long)m * SEQ + n);
      }
    }
}

// ---------------------------------------------------------------------------
// K3: row softmax in place over attn.  One 256-thread block per row of 2048.
// f32x4 loads, non-temporal f32x4 stores (attn >> L2).
// ---------------------------------------------------------------------------
__global__ __launch_bounds__(256) void mha_softmax_kernel(
    float* __restrict__ attn) {
  __shared__ float red[256];
  float* p = attn + (long)blockIdx.x * SEQ;
  const int tid = threadIdx.x;
  f32x4 v0 = *(const f32x4*)(p + tid * 8);
  f32x4 v1 = *(const f32x4*)(p + tid * 8 + 4);
  float mx = -INFINITY;
#pragma unroll
  for (int j = 0; j < 4; ++j) mx = fmaxf(mx, fmaxf(v0[j], v1[j]));
  red[tid] = mx;
  __syncthreads();
  for (int s = 128; s > 0; s >>= 1) {
    if (tid < s) red[tid] = fmaxf(red[tid], red[tid + s]);
    __syncthreads();
  }
  mx = red[0];
  __syncthreads();
  float sum = 0.0f;
#pragma unroll
  for (int j = 0; j < 4; ++j) {
    v0[j] = __expf(v0[j] - mx); sum += v0[j];
    v1[j] = __expf(v1[j] - mx); sum += v1[j];
  }
  red[tid] = sum;
  __syncthreads();
  for (int s = 128; s > 0; s >>= 1) {
    if (tid < s) red[tid] += red[tid + s];
    __syncthreads();
  }
  const float inv = 1.0f / red[0];
  v0 *= inv;
  v1 *= inv;
  __builtin_nontemporal_store(v0, (f32x4*)(p + tid * 8));
  __builtin_nontemporal_store(v1, (f32x4*)(p + tid * 8 + 4));
}

// ---------------------------------------------------------------------------
// K4: ctx = attn @ Vh per (b,h).  M=2048, N=64(=dk), K=2048.
// 256x64 block tile, 8 waves stacked in M; V (bf16) staged asynchronously,
// attn rows (f32) via register staging.  grid = (1, 8, 32).
// ctx stored merged-head: [b, s, h*64 + d] bf16.
// ---------------------------------------------------------------------------
__global__ __launch_bounds__(256) void mha_ctx_kernel(
    const float* __restrict__ attn, const bf16_t* __restrict__ Vt,
    bf16_t* __restrict__ ctx) {
  __shared__ bf16_t at[2][256 * LDSK];
  __shared__ bf16_t bt[2][64 * LDSK];
  const int bh = blockIdx.z;
  const int mBase = blockIdx.y * 256;
  const float*  aSrc = attn + (long)bh * SEQ * SEQ + (long)mBase * SEQ;
  const bf16_t* bSrc = Vt + (long)bh * DK * SEQ;
  const int tid = threadIdx.x;
  const int wave = tid >> 5, lane = tid & 31;
  const int waveM = wave * 32;

  v8f acc[2][4] = {};
  f32x4 aReg[8];
  tile_async_bf16<64>(bt[0], bSrc, SEQ, tid);
  tile_load_f32<256>(aReg, aSrc, SEQ, tid);
  tile_store_f32<256>(at[0], aReg, tid);
  WAIT_ASYNC();
  __syncthreads();
  int cur = 0;
  for (int k0 = 32; k0 < SEQ; k0 += 32) {
    tile_async_bf16<64>(bt[cur ^ 1], bSrc + k0, SEQ, tid);
    tile_load_f32<256>(aReg, aSrc + k0, SEQ, tid);
    SCHED_FENCE();   // async + staging loads issued before the WMMA block
    wave_mma_step(at[cur], bt[cur], waveM, 0, lane, acc);
    SCHED_FENCE();   // convert/ds_store only after the WMMA block
    tile_store_f32<256>(at[cur ^ 1], aReg, tid);
    WAIT_ASYNC();
    __syncthreads();
    cur ^= 1;
  }
  wave_mma_step(at[cur], bt[cur], waveM, 0, lane, acc);

  const int lr = lane & 15, hh = lane >> 4;
  const int batch = bh >> 4, h = bh & (NHEAD - 1);
#pragma unroll
  for (int mt = 0; mt < 2; ++mt)
#pragma unroll
    for (int nt = 0; nt < 4; ++nt) {
      F32Acc u; u.v = acc[mt][nt];
#pragma unroll
      for (int r = 0; r < 8; ++r) {
        const int s = mBase + waveM + mt * 16 + hh * 8 + r;
        const int d = nt * 16 + lr;
        ctx[((long)batch * SEQ + s) * DMODEL + h * DK + d] = (bf16_t)u.e[r];
      }
    }
}

// ---------------------------------------------------------------------------
// K5: out = ctx @ w_o^T + b_o.  M=4096, N=1024, K=1024.  grid = (8, 32).
// ctx (bf16) staged asynchronously; w_o (f32) via register staging.
// ---------------------------------------------------------------------------
__global__ __launch_bounds__(256) void mha_out_kernel(
    const bf16_t* __restrict__ ctx, const float* __restrict__ wo,
    const float* __restrict__ bo, float* __restrict__ out) {
  __shared__ bf16_t at[2][128 * LDSK];
  __shared__ bf16_t bt[2][128 * LDSK];
  const int mBase = blockIdx.y * 128;
  const int nBase = blockIdx.x * 128;
  const bf16_t* aSrc = ctx + (long)mBase * DMODEL;
  const float*  bSrc = wo + (long)nBase * DMODEL;
  const int tid = threadIdx.x;
  const int wave = tid >> 5, lane = tid & 31;
  const int waveM = (wave & 3) * 32, waveN = (wave >> 2) * 64;

  v8f acc[2][4] = {};
  f32x4 bReg[4];
  tile_async_bf16<128>(at[0], aSrc, DMODEL, tid);
  tile_load_f32<128>(bReg, bSrc, DMODEL, tid);
  tile_store_f32<128>(bt[0], bReg, tid);
  WAIT_ASYNC();
  __syncthreads();
  int cur = 0;
  for (int k0 = 32; k0 < DMODEL; k0 += 32) {
    tile_async_bf16<128>(at[cur ^ 1], aSrc + k0, DMODEL, tid);
    tile_load_f32<128>(bReg, bSrc + k0, DMODEL, tid);
    SCHED_FENCE();   // async + staging loads issued before the WMMA block
    wave_mma_step(at[cur], bt[cur], waveM, waveN, lane, acc);
    SCHED_FENCE();   // convert/ds_store only after the WMMA block
    tile_store_f32<128>(bt[cur ^ 1], bReg, tid);
    WAIT_ASYNC();
    __syncthreads();
    cur ^= 1;
  }
  wave_mma_step(at[cur], bt[cur], waveM, waveN, lane, acc);

  const int lr = lane & 15, hh = lane >> 4;
#pragma unroll
  for (int mt = 0; mt < 2; ++mt)
#pragma unroll
    for (int nt = 0; nt < 4; ++nt) {
      F32Acc u; u.v = acc[mt][nt];
#pragma unroll
      for (int r = 0; r < 8; ++r) {
        const int m = mBase + waveM + mt * 16 + hh * 8 + r;
        const int n = nBase + waveN + nt * 16 + lr;
        out[(long)m * DMODEL + n] = u.e[r] + bo[n];
      }
    }
}

// ---------------------------------------------------------------------------
extern "C" void kernel_launch(void* const* d_in, const int* in_sizes, int n_in,
                              void* d_out, int out_size, void* d_ws,
                              size_t ws_size, hipStream_t stream) {
  (void)in_sizes; (void)n_in; (void)out_size; (void)ws_size;
  const float* q  = (const float*)d_in[0];
  const float* k  = (const float*)d_in[1];
  const float* v  = (const float*)d_in[2];
  const float* wq = (const float*)d_in[3];
  const float* bq = (const float*)d_in[4];
  const float* wk = (const float*)d_in[5];
  const float* bk = (const float*)d_in[6];
  const float* wv = (const float*)d_in[7];
  const float* bv = (const float*)d_in[8];
  const float* wo = (const float*)d_in[9];
  const float* bo = (const float*)d_in[10];

  // Workspace layout (bf16): Qh | Kh | Vt | ctx  (8 MB each = 32 MB total)
  const long HSL = (long)BATCH * NHEAD * SEQ * DK;  // 4,194,304 elems
  bf16_t* Qh  = (bf16_t*)d_ws;
  bf16_t* Kh  = Qh + HSL;
  bf16_t* Vt  = Kh + HSL;
  bf16_t* ctxp = Vt + HSL;

  float* out  = (float*)d_out;
  float* attn = out + (long)BATCH * SEQ * DMODEL;  // + 4,194,304

  mha_proj_kernel<<<dim3(8, 32, 3), 256, 0, stream>>>(
      q, k, v, wq, bq, wk, bk, wv, bv, Qh, Kh, Vt);
  mha_scores_kernel<<<dim3(16, 16, 32), 256, 0, stream>>>(Qh, Kh, attn);
  mha_softmax_kernel<<<dim3(BATCH * NHEAD * SEQ), 256, 0, stream>>>(attn);
  mha_ctx_kernel<<<dim3(1, 8, 32), 256, 0, stream>>>(attn, Vt, ctxp);
  mha_out_kernel<<<dim3(8, 32, 1), 256, 0, stream>>>(ctxp, wo, bo, out);
}
